// NeurJudge_23390391894464
// MI455X (gfx1250) — matrix operand order repaired and checked by reading, not compiled, not run on hardware
//
#include <hip/hip_runtime.h>

typedef __attribute__((ext_vector_type(16))) _Float16 v16h;
typedef __attribute__((ext_vector_type(8)))  _Float16 v8h;
typedef __attribute__((ext_vector_type(8)))  float    v8f;

// ---------------------------------------------------------------------------
// WMMA fragment helpers (wave32, k order chosen by this code, same for A and B):
//  A lane m=lane&15, 16 contiguous K starting at k0 + (lane>=16 ? 16 : 0)
//  B lane n=lane&15, 16 contiguous K starting at k0 + (lane>=16 ? 16 : 0)
// ---------------------------------------------------------------------------
__device__ __forceinline__ v16h ldA16(const _Float16* p)
{
    return *(const v16h*)(p);
}
__device__ __forceinline__ v8f wmma16(v16h a, v16h b, v8f c)
{
    return __builtin_amdgcn_wmma_f32_16x16x32_f16(false, a, false, b,
                                                  (short)0, c, false, false);
}

// ---- CDNA5 async memory->LDS (ASYNCcnt) -----------------------------------
__device__ __forceinline__ void async_ld16(unsigned int ldsOff, const void* gaddr)
{
    asm volatile("global_load_async_to_lds_b128 %0, %1, off"
                 :: "v"(ldsOff), "v"((unsigned long long)(uintptr_t)gaddr)
                 : "memory");
}
__device__ __forceinline__ void wait_async0()
{
    asm volatile("s_wait_asynccnt 0" ::: "memory");
}

// ---- weight pack: f32 [N][K] -> f16 [Npad][Kpad], zero padded -------------
__global__ void k_pack_w(const float* __restrict__ W, _Float16* __restrict__ Wp,
                         int N, int K, int Npad, int Kpad)
{
    long long i = (long long)blockIdx.x * blockDim.x + threadIdx.x;
    if (i >= (long long)Npad * Kpad) return;
    int n = (int)(i / Kpad), k = (int)(i % Kpad);
    Wp[i] = (n < N && k < K) ? (_Float16)W[(size_t)n * K + k] : (_Float16)0.f;
}

// ---- embedding gather into GEMM layout: row = t*B + b, cols padded --------
__global__ void k_gather_emb(const int* __restrict__ tok, const float* __restrict__ embs,
                             _Float16* __restrict__ X, int Bn, int T, int D, int Dp)
{
    long long i = (long long)blockIdx.x * blockDim.x + threadIdx.x;
    if (i >= (long long)T * Bn * Dp) return;
    int row = (int)(i / Dp), d = (int)(i % Dp);
    int t = row / Bn, b = row % Bn;
    X[i] = (d < D) ? (_Float16)embs[(size_t)tok[(size_t)b * T + t] * D + d]
                   : (_Float16)0.f;
}

__global__ void k_zero_h(_Float16* __restrict__ p, long long n)
{
    long long i = (long long)blockIdx.x * blockDim.x + threadIdx.x;
    if (i < n) p[i] = (_Float16)0.f;
}

// ---- pack f32 activation slab [B,T,S] into f16 X[(t*B+b)][Kp] at colOff ---
__global__ void k_pack_x(const float* __restrict__ src, _Float16* __restrict__ X,
                         int Bn, int T, int S, int Kp, int colOff)
{
    long long i = (long long)blockIdx.x * blockDim.x + threadIdx.x;
    if (i >= (long long)T * Bn * S) return;
    int row = (int)(i / S), c = (int)(i % S);
    int t = row / Bn, b = row % Bn;
    X[(size_t)row * Kp + colOff + c] = (_Float16)src[((size_t)b * T + t) * S + c];
}

// ---------------------------------------------------------------------------
// Tiled WMMA GEMM, 2x2 register blocking (32x32 output per wave):
//   D[M][N] (f16) = A[M][Kp] * Bt[Np][Kp]^T + bias.   M multiple of 32,
//   Bt padded to Np rows (multiple of 32).  2 b128 loads per WMMA.
// ---------------------------------------------------------------------------
__global__ void k_gemm_gi(const _Float16* __restrict__ A, const _Float16* __restrict__ Bt,
                          const float* __restrict__ bias, _Float16* __restrict__ Dst,
                          int M, int N, int Kp)
{
    const int wid = threadIdx.x >> 5;
    const int tile = blockIdx.x * (blockDim.x >> 5) + wid;
    const int nT2 = (N + 31) >> 5;
    const int mT2 = M >> 5;
    if (tile >= nT2 * mT2) return;          // wave-uniform: EXEC stays all-ones
    const int m0 = (tile / nT2) * 32;
    const int n0 = (tile % nT2) * 32;
    const int lane = threadIdx.x & 31, ml = lane & 15, hi = lane >> 4;

    v8f c00 = {}, c01 = {}, c10 = {}, c11 = {};
    const _Float16* a0 = A  + ((size_t)m0 + ml) * Kp + hi * 16;
    const _Float16* a1 = a0 + (size_t)16 * Kp;
    const _Float16* b0 = Bt + ((size_t)n0 + ml) * Kp + hi * 16;
    const _Float16* b1 = b0 + (size_t)16 * Kp;
    for (int ks = 0; ks < (Kp >> 5); ++ks) {
        v16h A0 = ldA16(a0), A1 = ldA16(a1);
        v16h B0 = *(const v16h*)b0, B1 = *(const v16h*)b1;
        c00 = wmma16(A0, B0, c00);
        c01 = wmma16(A0, B1, c01);
        c10 = wmma16(A1, B0, c10);
        c11 = wmma16(A1, B1, c11);
        a0 += 32; a1 += 32; b0 += 32; b1 += 32;
    }
    for (int r = 0; r < 8; ++r) {
        int m  = m0 + r + hi * 8;            // M multiple of 32 -> always in range
        int na = n0 + ml, nb = n0 + 16 + ml;
        if (na < N) {
            Dst[(size_t)m * N + na]        = (_Float16)(c00[r] + (bias ? bias[na] : 0.f));
            Dst[(size_t)(m + 16) * N + na] = (_Float16)(c10[r] + (bias ? bias[na] : 0.f));
        }
        if (nb < N) {
            Dst[(size_t)m * N + nb]        = (_Float16)(c01[r] + (bias ? bias[nb] : 0.f));
            Dst[(size_t)(m + 16) * N + nb] = (_Float16)(c11[r] + (bias ? bias[nb] : 0.f));
        }
    }
}

// ---------------------------------------------------------------------------
// Persistent GRU scan. grid.x = 2 (dir 0 fwd, dir 1 bwd, concurrent).
// LDS layout (dynamic, offset 0): h f16 [32][Kp] | gh f16 [32][Np] |
//                                 gi stage f16 [32][N3] | ysum f32 [32*hid]
// Per step:
//   - issue async loads of gi[t] into LDS  (ASYNCcnt, hidden behind WMMA)
//   - phase G: gh = h @ Wh^T via 2x2 WMMA tiles -> LDS
//   - s_wait_asynccnt 0 + barrier
//   - phase U: GRU gate math, h update, y (or running mean) out
// ---------------------------------------------------------------------------
__global__ void k_gru_scan(const _Float16* __restrict__ giF, const _Float16* __restrict__ giB,
                           const _Float16* __restrict__ WhF, const _Float16* __restrict__ WhB,
                           const float* __restrict__ bhF, const float* __restrict__ bhB,
                           float* __restrict__ y,
                           int T, int hid, int N3, int Kp, int Np,
                           int ystride, int meanOnly)
{
    extern __shared__ char smem[];
    const unsigned off_gh = (unsigned)(32u * Kp * 2u);
    const unsigned off_gi = off_gh + (unsigned)(32u * Np * 2u);
    const unsigned off_ys = off_gi + (unsigned)(32u * N3 * 2u);
    _Float16* hS   = (_Float16*)smem;
    _Float16* ghS  = (_Float16*)(smem + off_gh);
    _Float16* giL  = (_Float16*)(smem + off_gi);
    float*    ysum = (float*)(smem + off_ys);

    const int dir = blockIdx.x;
    const _Float16* gi = dir ? giB : giF;
    const _Float16* Wh = dir ? WhB : WhF;
    const float*    bh = dir ? bhB : bhF;
    const int yoff = dir * hid;
    const int wid = threadIdx.x >> 5, nw = blockDim.x >> 5;
    const int lane = threadIdx.x & 31, ml = lane & 15, hi = lane >> 4;
    const unsigned stepChunks = (unsigned)(32 * N3 * 2) >> 4;   // 16B chunks per step

    for (int i = threadIdx.x; i < 32 * Kp; i += blockDim.x) hS[i] = (_Float16)0.f;
    if (meanOnly)
        for (int i = threadIdx.x; i < 32 * hid; i += blockDim.x) ysum[i] = 0.f;
    __syncthreads();

    for (int tt = 0; tt < T; ++tt) {
        const int t = dir ? (T - 1 - tt) : tt;
        // --- async stage of gi[t] into LDS (overlapped with WMMA below) ---
        {
            const char* gstep = (const char*)gi + (size_t)t * 32 * N3 * 2;
            for (unsigned c = threadIdx.x; c < stepChunks; c += blockDim.x)
                async_ld16(off_gi + (c << 4), gstep + ((size_t)c << 4));
        }
        // --- phase G: gh = h @ Wh^T (2x2 tiles, wave-uniform loop) ---
        for (int p = wid; p < (Np >> 5); p += nw) {
            const int n0 = p * 32;
            v8f c00 = {}, c01 = {}, c10 = {}, c11 = {};
            const _Float16* a0 = hS + (size_t)ml * Kp + hi * 16;
            const _Float16* a1 = a0 + (size_t)16 * Kp;
            const _Float16* b0 = Wh + ((size_t)n0 + ml) * Kp + hi * 16;
            const _Float16* b1 = b0 + (size_t)16 * Kp;
            for (int ks = 0; ks < (Kp >> 5); ++ks) {
                v16h A0 = ldA16(a0), A1 = ldA16(a1);
                v16h B0 = *(const v16h*)b0, B1 = *(const v16h*)b1;
                c00 = wmma16(A0, B0, c00);
                c01 = wmma16(A0, B1, c01);
                c10 = wmma16(A1, B0, c10);
                c11 = wmma16(A1, B1, c11);
                a0 += 32; a1 += 32; b0 += 32; b1 += 32;
            }
            for (int r = 0; r < 8; ++r) {
                int mr = r + hi * 8;
                int na = n0 + ml, nb = n0 + 16 + ml;
                ghS[(size_t)mr * Np + na]        = (_Float16)c00[r];
                ghS[(size_t)mr * Np + nb]        = (_Float16)c01[r];
                ghS[(size_t)(16 + mr) * Np + na] = (_Float16)c10[r];
                ghS[(size_t)(16 + mr) * Np + nb] = (_Float16)c11[r];
            }
        }
        wait_async0();
        __syncthreads();
        // --- phase U: gate math + state update ---
        for (int i = threadIdx.x; i < 32 * hid; i += blockDim.x) {
            const int b = i / hid, j = i % hid;
            const _Float16* g = giL + (size_t)b * N3;
            float ir = (float)g[j], iz = (float)g[hid + j], in_ = (float)g[2 * hid + j];
            float hr = (float)ghS[(size_t)b * Np + j]           + bh[j];
            float hz = (float)ghS[(size_t)b * Np + hid + j]     + bh[hid + j];
            float hn = (float)ghS[(size_t)b * Np + 2 * hid + j] + bh[2 * hid + j];
            float r = 1.f / (1.f + expf(-(ir + hr)));
            float z = 1.f / (1.f + expf(-(iz + hz)));
            float n = tanhf(in_ + r * hn);
            float hp = (float)hS[(size_t)b * Kp + j];
            float h2 = (1.f - z) * n + z * hp;
            hS[(size_t)b * Kp + j] = (_Float16)h2;
            if (meanOnly) ysum[i] += h2;
            else          y[((size_t)b * T + t) * ystride + yoff + j] = h2;
        }
        __syncthreads();
    }
    if (meanOnly)
        for (int i = threadIdx.x; i < 32 * hid; i += blockDim.x)
            y[(size_t)(i / hid) * ystride + yoff + (i % hid)] = ysum[i] / (float)T;
}

// ---- mean over T: [B,T,S] -> [B,S] ----------------------------------------
__global__ void k_mean_t(const float* __restrict__ y, float* __restrict__ out,
                         int Bn, int T, int S)
{
    int i = blockIdx.x * blockDim.x + threadIdx.x;
    if (i >= Bn * S) return;
    int b = i / S, d = i % S;
    float s = 0.f;
    for (int t = 0; t < T; ++t) s += y[((size_t)b * T + t) * S + d];
    out[i] = s / (float)T;
}

// ---- mask_attention + fact_separation (one block per (b,l)) ---------------
__global__ void k_attn_sep(const float* __restrict__ ctx, const float* __restrict__ q,
                           float* __restrict__ similar, float* __restrict__ secOut,
                           int L, int LV, int Dm)
{
    const int b = blockIdx.x / L, l = blockIdx.x % L;
    const float* cv = ctx + ((size_t)b * L + l) * Dm;
    const float* qb = q + (size_t)b * LV * Dm;
    __shared__ float aw[112];
    __shared__ float sc[304];
    __shared__ float r3[128], r4[128];
    const int tid = threadIdx.x;

    for (int v = tid; v < LV; v += blockDim.x) {
        const float* qv = qb + (size_t)v * Dm;
        float s = 0.f;
        for (int d = 0; d < Dm; ++d) s += cv[d] * qv[d];
        aw[v] = s;
    }
    __syncthreads();
    if (tid == 0) {   // masked softmax over v (LV=100, trivial serial cost)
        float mx = -__builtin_inff();
        for (int v = 0; v < LV; ++v) {
            float m = (aw[v] == 0.f) ? -__builtin_inff() : aw[v];
            aw[v] = m; if (m > mx) mx = m;
        }
        if (mx == -__builtin_inff()) {
            for (int v = 0; v < LV; ++v) aw[v] = 0.f;   // NaN -> 0 path
        } else {
            float se = 0.f;
            for (int v = 0; v < LV; ++v) { float e = expf(aw[v] - mx); aw[v] = e; se += e; }
            float inv = 1.f / se;
            for (int v = 0; v < LV; ++v) aw[v] *= inv;
        }
    }
    __syncthreads();
    for (int d = tid; d < Dm; d += blockDim.x) {
        float s = 0.f;
        for (int v = 0; v < LV; ++v) s += aw[v] * qb[(size_t)v * Dm + d];
        sc[d] = s;
    }
    __syncthreads();
    float p3 = 0.f, p4 = 0.f;
    for (int d = tid; d < Dm; d += blockDim.x) { float sd = sc[d]; p3 += cv[d] * sd; p4 += sd * sd; }
    r3[tid] = p3; r4[tid] = p4;
    __syncthreads();
    for (int off = blockDim.x >> 1; off > 0; off >>= 1) {
        if (tid < off) { r3[tid] += r3[tid + off]; r4[tid] += r4[tid + off]; }
        __syncthreads();
    }
    float coef = r3[0] / (r4[0] + 1e-10f);
    for (int d = tid; d < Dm; d += blockDim.x) {
        size_t idx = ((size_t)b * L + l) * Dm + d;
        float sim = coef * sc[d];
        similar[idx] = sim;
        secOut[idx] = cv[d] - sim;
    }
}

// ---- classifier head: logits + log-softmax NLL + argmax (block per b) -----
__global__ void k_head(const float* __restrict__ x, const float* __restrict__ W,
                       const float* __restrict__ bias, const int* __restrict__ labels,
                       float* __restrict__ pred, float* __restrict__ lossBuf,
                       int F, int C)
{
    const int b = blockIdx.x, tid = threadIdx.x;
    __shared__ float lg[128];
    const float* xb = x + (size_t)b * F;
    for (int c = tid; c < C; c += blockDim.x) {
        const float* w = W + (size_t)c * F;
        float s = bias[c];
        for (int f = 0; f < F; ++f) s += xb[f] * w[f];
        lg[c] = s;
    }
    __syncthreads();
    if (tid == 0) {
        float mx = lg[0]; int am = 0;
        for (int c = 1; c < C; ++c) if (lg[c] > mx) { mx = lg[c]; am = c; }
        float se = 0.f;
        for (int c = 0; c < C; ++c) se += expf(lg[c] - mx);
        float lse = mx + logf(se);
        lossBuf[b] = -(lg[labels[b]] - lse);
        pred[b] = (float)am;
    }
}

__global__ void k_loss_mean(const float* __restrict__ lossBuf, float* __restrict__ out, int Bn)
{
    if (threadIdx.x == 0 && blockIdx.x == 0) {
        float s = 0.f;
        for (int b = 0; b < Bn; ++b) s += lossBuf[b];
        *out = s / (float)Bn;
    }
}

// ===========================================================================
extern "C" void kernel_launch(void* const* d_in, const int* in_sizes, int n_in,
                              void* d_out, int out_size, void* d_ws, size_t ws_size,
                              hipStream_t stream)
{
    (void)in_sizes; (void)n_in; (void)out_size; (void)ws_size;
    const int B = 32, L = 512, LV = 100, D = 200;
    const int Dp = 224;                                   // K pad (mult of 32)
    const int hidE = 150, N3E = 450, KpEh = 160, NpE = 480;   // Np mult of 32
    const int KinA = 600, hidA = 300, N3A = 900, KpA = 608, KpAh = 320, NpA = 928;
    const int KinT = 900, hidT = 450, N3T = 1350, KpT = 928, KpTh = 480, NpT = 1376;
    const int Mdoc = B * L, Mv = B * LV;

    const int* documents       = (const int*)d_in[0];
    const int* verdict_charge  = (const int*)d_in[1];
    const int* verdict_article = (const int*)d_in[2];
    const int* accu_labels     = (const int*)d_in[3];
    const int* law_labels      = (const int*)d_in[4];
    const int* term_labels     = (const int*)d_in[5];
    const float* embs          = (const float*)d_in[6];
    const float* eWi_f = (const float*)d_in[7];  const float* eWh_f = (const float*)d_in[8];
    const float* ebi_f = (const float*)d_in[9];  const float* ebh_f = (const float*)d_in[10];
    const float* eWi_b = (const float*)d_in[11]; const float* eWh_b = (const float*)d_in[12];
    const float* ebi_b = (const float*)d_in[13]; const float* ebh_b = (const float*)d_in[14];
    const float* aWi_f = (const float*)d_in[15]; const float* aWh_f = (const float*)d_in[16];
    const float* abi_f = (const float*)d_in[17]; const float* abh_f = (const float*)d_in[18];
    const float* aWi_b = (const float*)d_in[19]; const float* aWh_b = (const float*)d_in[20];
    const float* abi_b = (const float*)d_in[21]; const float* abh_b = (const float*)d_in[22];
    const float* tWi_f = (const float*)d_in[23]; const float* tWh_f = (const float*)d_in[24];
    const float* tbi_f = (const float*)d_in[25]; const float* tbh_f = (const float*)d_in[26];
    const float* tWi_b = (const float*)d_in[27]; const float* tWh_b = (const float*)d_in[28];
    const float* tbi_b = (const float*)d_in[29]; const float* tbh_b = (const float*)d_in[30];
    const float* charge_W  = (const float*)d_in[31]; const float* charge_b  = (const float*)d_in[32];
    const float* article_W = (const float*)d_in[33]; const float* article_b = (const float*)d_in[34];
    const float* time_W    = (const float*)d_in[35]; const float* time_b    = (const float*)d_in[36];

    char* base = (char*)d_ws;
    size_t cur = 0;
    auto alloc = [&](size_t bytes) -> void* {
        void* p = base + cur; cur = (cur + bytes + 255) & ~(size_t)255; return p;
    };
    // persistent f16 packed weights
    _Float16* WiEf = (_Float16*)alloc((size_t)NpE * Dp * 2);
    _Float16* WiEb = (_Float16*)alloc((size_t)NpE * Dp * 2);
    _Float16* WhEf = (_Float16*)alloc((size_t)NpE * KpEh * 2);
    _Float16* WhEb = (_Float16*)alloc((size_t)NpE * KpEh * 2);
    _Float16* WiAf = (_Float16*)alloc((size_t)NpA * KpA * 2);
    _Float16* WiAb = (_Float16*)alloc((size_t)NpA * KpA * 2);
    _Float16* WhAf = (_Float16*)alloc((size_t)NpA * KpAh * 2);
    _Float16* WhAb = (_Float16*)alloc((size_t)NpA * KpAh * 2);
    _Float16* WiTf = (_Float16*)alloc((size_t)NpT * KpT * 2);
    _Float16* WiTb = (_Float16*)alloc((size_t)NpT * KpT * 2);
    _Float16* WhTf = (_Float16*)alloc((size_t)NpT * KpTh * 2);
    _Float16* WhTb = (_Float16*)alloc((size_t)NpT * KpTh * 2);
    // persistent activations
    float* dhid = (float*)alloc((size_t)B * L * 300 * 4);
    float* vhid = (float*)alloc((size_t)B * LV * 300 * 4);
    float* adc  = (float*)alloc((size_t)B * L * 300 * 4);
    float* sep  = (float*)alloc((size_t)B * L * 300 * 4);   // "sec"
    float* ssc  = (float*)alloc((size_t)B * L * 300 * 4);
    float* dsc  = (float*)alloc((size_t)B * L * 300 * 4);
    _Float16* Xv   = (_Float16*)alloc((size_t)Mv * Dp * 2);
    _Float16* giVf = (_Float16*)alloc((size_t)Mv * N3E * 2);
    _Float16* giVb = (_Float16*)alloc((size_t)Mv * N3E * 2);
    float* famean  = (float*)alloc((size_t)32 * 600 * 4);
    float* ftmean  = (float*)alloc((size_t)32 * 900 * 4);
    float* lossBuf = (float*)alloc(32 * 4);
    const size_t arena = cur;   // stage-local scratch reuses this region
    auto aralloc = [&](size_t& c, size_t bytes) -> void* {
        void* p = base + c; c = (c + bytes + 255) & ~(size_t)255; return p;
    };

    auto packw = [&](const float* W, _Float16* Wp, int N, int K, int Np_, int Kp_) {
        long long n = (long long)Np_ * Kp_;
        k_pack_w<<<(unsigned)((n + 255) / 256), 256, 0, stream>>>(W, Wp, N, K, Np_, Kp_);
    };
    auto gemm = [&](const _Float16* A_, const _Float16* Bt, const float* bias,
                    _Float16* Dst, int M, int N, int Kp_) {
        int tiles = (M >> 5) * ((N + 31) >> 5);
        k_gemm_gi<<<(unsigned)((tiles + 7) / 8), 256, 0, stream>>>(A_, Bt, bias, Dst, M, N, Kp_);
    };
    auto scan = [&](const _Float16* gF, const _Float16* gB,
                    const _Float16* WF, const _Float16* WB,
                    const float* bF, const float* bB, float* y,
                    int T, int hid, int N3, int Kp_, int Np_, int ystride, int meanOnly) {
        size_t lds = (size_t)32 * Kp_ * 2 + (size_t)32 * Np_ * 2 + (size_t)32 * N3 * 2
                   + (meanOnly ? (size_t)32 * hid * 4 : 0);
        k_gru_scan<<<2, 256, lds, stream>>>(gF, gB, WF, WB, bF, bB, y,
                                            T, hid, N3, Kp_, Np_, ystride, meanOnly);
    };

    // --- pack all weights to f16 (padded) ---
    packw(eWi_f, WiEf, N3E, D, NpE, Dp);      packw(eWi_b, WiEb, N3E, D, NpE, Dp);
    packw(eWh_f, WhEf, N3E, hidE, NpE, KpEh); packw(eWh_b, WhEb, N3E, hidE, NpE, KpEh);
    packw(aWi_f, WiAf, N3A, KinA, NpA, KpA);  packw(aWi_b, WiAb, N3A, KinA, NpA, KpA);
    packw(aWh_f, WhAf, N3A, hidA, NpA, KpAh); packw(aWh_b, WhAb, N3A, hidA, NpA, KpAh);
    packw(tWi_f, WiTf, N3T, KinT, NpT, KpT);  packw(tWi_b, WiTb, N3T, KinT, NpT, KpT);
    packw(tWh_f, WhTf, N3T, hidT, NpT, KpTh); packw(tWh_b, WhTb, N3T, hidT, NpT, KpTh);

    float* out = (float*)d_out;
    float* predA = out;       float* predL = out + 32;  float* predT = out + 64;
    float* lossA = out + 96;  float* lossL = out + 97;  float* lossT = out + 98;
    float* df    = out + 99;  // [32,300]

    // --- stage 1: encoder biGRU over documents -> dhid [B,L,300] ---
    {
        size_t ac = arena;
        _Float16* Xdoc = (_Float16*)aralloc(ac, (size_t)Mdoc * Dp * 2);
        _Float16* giEf = (_Float16*)aralloc(ac, (size_t)Mdoc * N3E * 2);
        _Float16* giEb = (_Float16*)aralloc(ac, (size_t)Mdoc * N3E * 2);
        long long n = (long long)Mdoc * Dp;
        k_gather_emb<<<(unsigned)((n + 255) / 256), 256, 0, stream>>>(documents, embs, Xdoc, B, L, D, Dp);
        gemm(Xdoc, WiEf, ebi_f, giEf, Mdoc, N3E, Dp);
        gemm(Xdoc, WiEb, ebi_b, giEb, Mdoc, N3E, Dp);
        scan(giEf, giEb, WhEf, WhEb, ebh_f, ebh_b, dhid, L, hidE, N3E, KpEh, NpE, 300, 0);
    }
    // df + accusation head
    k_mean_t<<<(32 * 300 + 255) / 256, 256, 0, stream>>>(dhid, df, 32, L, 300);
    k_head<<<32, 128, 0, stream>>>(df, charge_W, charge_b, accu_labels, predA, lossBuf, 300, 119);
    k_loss_mean<<<1, 1, 0, stream>>>(lossBuf, lossA, 32);

    // --- fact_separation(verdict_charge, dhid) -> adc, sep ---
    {
        long long n = (long long)Mv * Dp;
        k_gather_emb<<<(unsigned)((n + 255) / 256), 256, 0, stream>>>(verdict_charge, embs, Xv, B, LV, D, Dp);
        gemm(Xv, WiEf, ebi_f, giVf, Mv, N3E, Dp);
        gemm(Xv, WiEb, ebi_b, giVb, Mv, N3E, Dp);
        scan(giVf, giVb, WhEf, WhEb, ebh_f, ebh_b, vhid, LV, hidE, N3E, KpEh, NpE, 300, 0);
        k_attn_sep<<<B * L, 128, 0, stream>>>(dhid, vhid, adc, sep, L, LV, 300);
    }

    // --- stage 2: article biGRU over concat([dhid, adc]) -> mean -> law head ---
    {
        size_t ac = arena;
        _Float16* XA   = (_Float16*)aralloc(ac, (size_t)Mdoc * KpA * 2);
        _Float16* giAf = (_Float16*)aralloc(ac, (size_t)Mdoc * N3A * 2);
        _Float16* giAb = (_Float16*)aralloc(ac, (size_t)Mdoc * N3A * 2);
        long long nz = (long long)Mdoc * KpA;
        k_zero_h<<<(unsigned)((nz + 255) / 256), 256, 0, stream>>>(XA, nz);
        long long np = (long long)Mdoc * 300;
        k_pack_x<<<(unsigned)((np + 255) / 256), 256, 0, stream>>>(dhid, XA, B, L, 300, KpA, 0);
        k_pack_x<<<(unsigned)((np + 255) / 256), 256, 0, stream>>>(adc,  XA, B, L, 300, KpA, 300);
        gemm(XA, WiAf, abi_f, giAf, Mdoc, N3A, KpA);
        gemm(XA, WiAb, abi_b, giAb, Mdoc, N3A, KpA);
        scan(giAf, giAb, WhAf, WhAb, abh_f, abh_b, famean, L, hidA, N3A, KpAh, NpA, 600, 1);
        k_head<<<32, 128, 0, stream>>>(famean, article_W, article_b, law_labels, predL, lossBuf, 600, 103);
        k_loss_mean<<<1, 1, 0, stream>>>(lossBuf, lossL, 32);
    }

    // --- fact_separation(verdict_article, sep) -> ssc, dsc ---
    {
        long long n = (long long)Mv * Dp;
        k_gather_emb<<<(unsigned)((n + 255) / 256), 256, 0, stream>>>(verdict_article, embs, Xv, B, LV, D, Dp);
        gemm(Xv, WiEf, ebi_f, giVf, Mv, N3E, Dp);
        gemm(Xv, WiEb, ebi_b, giVb, Mv, N3E, Dp);
        scan(giVf, giVb, WhEf, WhEb, ebh_f, ebh_b, vhid, LV, hidE, N3E, KpEh, NpE, 300, 0);
        k_attn_sep<<<B * L, 128, 0, stream>>>(sep, vhid, ssc, dsc, L, LV, 300);
    }

    // --- stage 3: term biGRU over concat([dhid, ssc, dsc]) -> mean -> term head ---
    {
        size_t ac = arena;
        _Float16* XT   = (_Float16*)aralloc(ac, (size_t)Mdoc * KpT * 2);
        _Float16* giTf = (_Float16*)aralloc(ac, (size_t)Mdoc * N3T * 2);
        _Float16* giTb = (_Float16*)aralloc(ac, (size_t)Mdoc * N3T * 2);
        long long nz = (long long)Mdoc * KpT;
        k_zero_h<<<(unsigned)((nz + 255) / 256), 256, 0, stream>>>(XT, nz);
        long long np = (long long)Mdoc * 300;
        k_pack_x<<<(unsigned)((np + 255) / 256), 256, 0, stream>>>(dhid, XT, B, L, 300, KpT, 0);
        k_pack_x<<<(unsigned)((np + 255) / 256), 256, 0, stream>>>(ssc,  XT, B, L, 300, KpT, 300);
        k_pack_x<<<(unsigned)((np + 255) / 256), 256, 0, stream>>>(dsc,  XT, B, L, 300, KpT, 600);
        gemm(XT, WiTf, tbi_f, giTf, Mdoc, N3T, KpT);
        gemm(XT, WiTb, tbi_b, giTb, Mdoc, N3T, KpT);
        scan(giTf, giTb, WhTf, WhTb, tbh_f, tbh_b, ftmean, L, hidT, N3T, KpTh, NpT, 900, 1);
        k_head<<<32, 128, 0, stream>>>(ftmean, time_W, time_b, term_labels, predT, lossBuf, 900, 11);
        k_loss_mean<<<1, 1, 0, stream>>>(lossBuf, lossT, 32);
    }
}
